// Attention_81209241633081
// MI455X (gfx1250) — matrix operand order repaired and checked by reading, not compiled, not run on hardware
//
#include <hip/hip_runtime.h>

// ---------------------------------------------------------------------------
// Causal multi-head attention block for MI455X (gfx1250, wave32, WMMA + TDM).
// B=4, T=2048, C=1024, H=16, D=64.  All matmuls via v_wmma_f32_16x16x32_bf16.
// V is stored transposed [B,H,D,T] so P@V B-fragments are contiguous loads.
// out_gemm stages its bf16 A tile via the Tensor Data Mover (tensor_load_to_lds).
// ---------------------------------------------------------------------------

typedef __attribute__((ext_vector_type(16))) __bf16 v16bf;
typedef __attribute__((ext_vector_type(8)))  __bf16 v8bf;
typedef __attribute__((ext_vector_type(8)))  float  v8f;
typedef __attribute__((ext_vector_type(4)))  unsigned int u32x4;
typedef __attribute__((ext_vector_type(4)))  int          i32x4;
typedef __attribute__((ext_vector_type(8)))  int          i32x8;

#define WMMA_BF16(a, b, c) \
  __builtin_amdgcn_wmma_f32_16x16x32_bf16(false, (a), false, (b), (short)0, (c), false, false)

#if __has_builtin(__builtin_amdgcn_tensor_load_to_lds) && \
    __has_builtin(__builtin_amdgcn_s_wait_tensorcnt)
#define HAVE_TDM 1
#else
#define HAVE_TDM 0
#endif

constexpr unsigned Bn = 4, Tn = 2048, Cn = 1024, Hn = 16, Dn = 64;
constexpr unsigned Mrows = Bn * Tn;          // 8192
constexpr unsigned N_QKV = 3 * Cn;           // 3072

// A-fragment (16x32 bf16): lanes 0-15 hold K [0..15];
// lanes 16-31 hold K [16..31].  `kchunk` = 0 or 8 per lane half.
__device__ __forceinline__ v16bf make_afrag(const __bf16* row, int kchunk) {
  v8bf lo = *(const v8bf*)(row + 2 * kchunk);
  v8bf hi = *(const v8bf*)(row + 2 * kchunk + 8);
  v16bf a;
#pragma unroll
  for (int i = 0; i < 8; ++i) { a[i] = lo[i]; a[i + 8] = hi[i]; }
  return a;
}

// B-fragment (32x16 bf16): lane holds column N=lane%16, 16 contiguous K
// elements starting at K = 16*(lane/16).  `p` points at those 16 values.
__device__ __forceinline__ v16bf make_bfrag(const __bf16* p) {
  v8bf lo = *(const v8bf*)p;
  v8bf hi = *(const v8bf*)(p + 8);
  v16bf b;
#pragma unroll
  for (int i = 0; i < 8; ++i) { b[i] = lo[i]; b[i + 8] = hi[i]; }
  return b;
}

// ---------------------------------------------------------------------------
// Kernel 1: QKV projection.  x[8192,1024] fp32 @ w[1024,3072] fp32 ->
// Q (pre-scaled by 1/sqrt(D)), K in [B,H,T,D]; V transposed in [B,H,D,T].
// Block = 256 threads (8 waves).  Block tile 128(M) x 128(N), wave 32x64.
// ---------------------------------------------------------------------------
__global__ __launch_bounds__(256)
void qkv_gemm(const float* __restrict__ x, const float* __restrict__ w,
              __bf16* __restrict__ qb, __bf16* __restrict__ kb,
              __bf16* __restrict__ vt) {
  __shared__ __bf16 As[128][40];   // [m][k]
  __shared__ __bf16 Bs[128][40];   // transposed [n][k]

  const unsigned tid  = threadIdx.x;
  const unsigned lane = tid & 31, wave = tid >> 5;
  const unsigned nloc = lane & 15, half = lane >> 4;
  const int kchunk = half ? 8 : 0;
  const unsigned mg = wave >> 1;            // 0..3 : 32-row group
  const unsigned ng = wave & 1;             // 0..1 : 64-col group
  const unsigned n0 = blockIdx.x * 128;
  const unsigned m0 = blockIdx.y * 128;

  v8f acc[2][4] = {};
  for (unsigned k0 = 0; k0 < Cn; k0 += 32) {
    // stage A (128x32 fp32 -> bf16)
    for (unsigned idx = tid; idx < 128 * 8; idx += 256) {
      unsigned r = idx >> 3, c = (idx & 7) * 4;
      const float* src = x + (size_t)(m0 + r) * Cn + k0 + c;
      const float4 f = *(const float4*)src;
      __builtin_prefetch(src + 32, 0, 3);              // next k-tile
      As[r][c + 0] = (__bf16)f.x; As[r][c + 1] = (__bf16)f.y;
      As[r][c + 2] = (__bf16)f.z; As[r][c + 3] = (__bf16)f.w;
    }
    // stage B transposed (32x128 fp32 -> bf16 [n][k])
    for (unsigned idx = tid; idx < 32 * 32; idx += 256) {
      unsigned r = idx >> 5, c = (idx & 31) * 4;
      const float* src = w + (size_t)(k0 + r) * N_QKV + n0 + c;
      const float4 f = *(const float4*)src;
      __builtin_prefetch(src + (size_t)32 * N_QKV, 0, 3);
      Bs[c + 0][r] = (__bf16)f.x; Bs[c + 1][r] = (__bf16)f.y;
      Bs[c + 2][r] = (__bf16)f.z; Bs[c + 3][r] = (__bf16)f.w;
    }
    __syncthreads();

    const v16bf a0 = make_afrag(&As[mg * 32 + nloc][0], kchunk);
    const v16bf a1 = make_afrag(&As[mg * 32 + 16 + nloc][0], kchunk);
#pragma unroll
    for (int j = 0; j < 4; ++j) {
      const v16bf b = make_bfrag(&Bs[ng * 64 + j * 16 + nloc][half << 4]);
      acc[0][j] = WMMA_BF16(a0, b, acc[0][j]);
      acc[1][j] = WMMA_BF16(a1, b, acc[1][j]);
    }
    __syncthreads();
  }

  // epilogue: Q scaled, K row-major [B,H,T,D]; V transposed [B,H,D,T]
#pragma unroll
  for (unsigned j = 0; j < 4; ++j) {
    const unsigned n = n0 + ng * 64 + j * 16 + nloc;
    const unsigned which = n / Cn, c = n % Cn;
    const unsigned head = c / Dn, d = c % Dn;
#pragma unroll
    for (unsigned i = 0; i < 2; ++i) {
      const unsigned mbase = m0 + mg * 32 + i * 16 + 8 * half;
      const unsigned bI = mbase / Tn;            // 128-row tiles never straddle b
      const unsigned t0 = mbase % Tn;
      if (which == 2) {                          // V: contiguous along t
        v8bf pack;
#pragma unroll
        for (int r = 0; r < 8; ++r) pack[r] = (__bf16)acc[i][j][r];
        *(v8bf*)(vt + (((size_t)(bI * Hn + head)) * Dn + d) * Tn + t0) = pack;
      } else {
        __bf16* dst = (which == 0) ? qb : kb;
        const float scale = (which == 0) ? 0.125f : 1.0f;
#pragma unroll
        for (unsigned r = 0; r < 8; ++r)
          dst[(((size_t)(bI * Hn + head)) * Tn + t0 + r) * Dn + d] =
              (__bf16)(acc[i][j][r] * scale);
      }
    }
  }
}

// ---------------------------------------------------------------------------
// Kernel 2: flash attention.  Each wave owns one 16-row query tile; key loop
// in blocks of 32 with online softmax.  8 waves/block, grid = (T/128, B*H).
// No block barriers: per-wave causal trip counts differ.
// ---------------------------------------------------------------------------
__global__ __launch_bounds__(256)
void attn(const __bf16* __restrict__ qb, const __bf16* __restrict__ kb,
          const __bf16* __restrict__ vt, __bf16* __restrict__ y) {
  __shared__ __bf16 Pl[8][16][40];   // per-wave P tile, [m][s]

  const unsigned tid  = threadIdx.x;
  const unsigned lane = tid & 31, wave = tid >> 5;
  const unsigned nloc = lane & 15, half = lane >> 4;
  const int kchunk = half ? 8 : 0;
  const unsigned bh   = blockIdx.y;               // b*H + h
  const unsigned bI   = bh >> 4, head = bh & 15;
  const unsigned q0   = (blockIdx.x * 8 + wave) * 16;

  const __bf16* Q = qb + (size_t)bh * Tn * Dn;
  const __bf16* K = kb + (size_t)bh * Tn * Dn;
  const __bf16* V = vt + (size_t)bh * Dn * Tn;   // [d][t]

  // Q fragments (already pre-scaled): A 16x32 for d=[0,32) and d=[32,64)
  const __bf16* qrow = Q + (size_t)(q0 + nloc) * Dn;
  const v16bf qa0 = make_afrag(qrow + 0,  kchunk);
  const v16bf qa1 = make_afrag(qrow + 32, kchunk);

  float mrow[8], lrow[8];
  v8f o0 = {}, o1 = {}, o2 = {}, o3 = {};
#pragma unroll
  for (int r = 0; r < 8; ++r) { mrow[r] = -__builtin_inff(); lrow[r] = 0.f; }

  for (unsigned s0 = 0; s0 < q0 + 16; s0 += 32) {
    // prefetch next key/value block
    __builtin_prefetch(K + (size_t)(s0 + 32 + nloc) * Dn, 0, 3);
    __builtin_prefetch(V + (size_t)nloc * Tn + s0 + 32, 0, 3);

    // ---- S = Q @ K^T : two 16x16 f32 tiles (keys s0..+15, s0+16..+31) ----
    v8f c0 = {}, c1 = {};
    {
      const __bf16* krA = K + (size_t)(s0 + nloc)      * Dn + (half << 4);
      const __bf16* krB = K + (size_t)(s0 + 16 + nloc) * Dn + (half << 4);
      c0 = WMMA_BF16(qa0, make_bfrag(krA),      c0);
      c0 = WMMA_BF16(qa1, make_bfrag(krA + 32), c0);
      c1 = WMMA_BF16(qa0, make_bfrag(krB),      c1);
      c1 = WMMA_BF16(qa1, make_bfrag(krB + 32), c1);
    }

    // ---- causal mask + online softmax (rows live across 16-lane halves) --
#pragma unroll
    for (int r = 0; r < 8; ++r) {
      const int qi = (int)(q0 + r + 8 * half);
      float s0v = ((int)(s0 + nloc)      <= qi) ? c0[r] : -__builtin_inff();
      float s1v = ((int)(s0 + 16 + nloc) <= qi) ? c1[r] : -__builtin_inff();
      float rmax = fmaxf(s0v, s1v);
      rmax = fmaxf(rmax, __shfl_xor(rmax, 1));
      rmax = fmaxf(rmax, __shfl_xor(rmax, 2));
      rmax = fmaxf(rmax, __shfl_xor(rmax, 4));
      rmax = fmaxf(rmax, __shfl_xor(rmax, 8));
      const float mnew  = fmaxf(mrow[r], rmax);
      const float alpha = __expf(mrow[r] - mnew);
      const float p0 = __expf(s0v - mnew);
      const float p1 = __expf(s1v - mnew);
      float rsum = p0 + p1;
      rsum += __shfl_xor(rsum, 1);
      rsum += __shfl_xor(rsum, 2);
      rsum += __shfl_xor(rsum, 4);
      rsum += __shfl_xor(rsum, 8);
      mrow[r] = mnew;
      lrow[r] = lrow[r] * alpha + rsum;
      o0[r] *= alpha; o1[r] *= alpha; o2[r] *= alpha; o3[r] *= alpha;
      const unsigned m = r + 8 * half;
      Pl[wave][m][nloc]      = (__bf16)p0;   // C-layout -> LDS (re-layout)
      Pl[wave][m][nloc + 16] = (__bf16)p1;
    }

    // ---- O += P @ V  (P: 16x32 A-frag from LDS; V^T: contiguous B-frags) -
    const v16bf pa = make_afrag(&Pl[wave][nloc][0], kchunk);
#pragma unroll
    for (int j = 0; j < 4; ++j) {
      const v16bf bv =
          make_bfrag(V + (size_t)(j * 16 + nloc) * Tn + s0 + (half << 4));
      v8f& o = (j == 0) ? o0 : (j == 1) ? o1 : (j == 2) ? o2 : o3;
      o = WMMA_BF16(pa, bv, o);
    }
  }

  // ---- normalize and store y[b][t][h*D + d] as bf16 -----------------------
#pragma unroll
  for (int r = 0; r < 8; ++r) {
    const float inv = 1.0f / lrow[r];
    const unsigned t = q0 + r + 8 * half;
    __bf16* yp = y + ((size_t)bI * Tn + t) * Cn + head * Dn + nloc;
    yp[0]  = (__bf16)(o0[r] * inv);
    yp[16] = (__bf16)(o1[r] * inv);
    yp[32] = (__bf16)(o2[r] * inv);
    yp[48] = (__bf16)(o3[r] * inv);
  }
}

// ---------------------------------------------------------------------------
// Kernel 3: output projection.  y[8192,1024] bf16 @ w_proj[1024,1024] fp32
// + bias -> out fp32.  A tile staged into LDS by the Tensor Data Mover:
// 128x32 bf16 tile, LDS padding 4 DWORDs after every 16 DWORDs (row = 40
// bf16 = 20 DWORDs, matching As[128][40]).
// ---------------------------------------------------------------------------
__global__ __launch_bounds__(256)
void out_gemm(const __bf16* __restrict__ y, const float* __restrict__ w,
              const float* __restrict__ bias, float* __restrict__ out) {
  __shared__ __bf16 As[128][40];
  __shared__ __bf16 Bs[128][40];

  const unsigned tid  = threadIdx.x;
  const unsigned lane = tid & 31, wave = tid >> 5;
  const unsigned nloc = lane & 15, half = lane >> 4;
  const int kchunk = half ? 8 : 0;
  const unsigned mg = wave >> 1, ng = wave & 1;
  const unsigned n0 = blockIdx.x * 128;
  const unsigned m0 = blockIdx.y * 128;

  v8f acc[2][4] = {};
  for (unsigned k0 = 0; k0 < Cn; k0 += 32) {
#if HAVE_TDM
    // ---- TDM: DMA the 128x32 bf16 A tile global -> LDS (wave 0 issues) ----
    if (wave == 0) {
      const unsigned lds_addr =
          (unsigned)(unsigned long long)(const void*)&As[0][0];
      const unsigned long long ga =
          (unsigned long long)(const void*)(y + (size_t)m0 * Cn + k0);
      u32x4 g0;
      g0[0] = 1u;                                        // count=1, user mode
      g0[1] = lds_addr;                                  // lds_addr [63:32]
      g0[2] = (unsigned)(ga & 0xffffffffu);              // global_addr lo
      g0[3] = (unsigned)((ga >> 32) & 0x01ffffffu) | (2u << 30);  // hi | type=2
      i32x8 g1;
      g1[0] = (int)((1u << 16)    // data_size = 2 bytes
                  | (1u << 20)    // pad_enable
                  | (3u << 22)    // pad_interval: 16 DWORDs
                  | (3u << 25));  // pad_amount: 4 DWORDs
      g1[1] = (int)((Cn & 0xffffu) << 16);     // tensor_dim0 = 1024 (lo16)
      g1[2] = (int)((Mrows & 0xffffu) << 16);  // dim0 hi=0 | tensor_dim1 lo16
      g1[3] = (int)(32u << 16);                // dim1 hi=0 | tile_dim0 = 32
      g1[4] = 128;                             // tile_dim1 = 128, tile_dim2 = 0
      g1[5] = (int)Cn;                         // tensor_dim0_stride = 1024
      g1[6] = 0;
      g1[7] = 0;
      __builtin_amdgcn_tensor_load_to_lds(g0, g1, (i32x4)0, (i32x4)0,
                                          (i32x8)0, 0);
    }
#else
    for (unsigned idx = tid; idx < 128 * 4; idx += 256) {
      unsigned r = idx >> 2, c = (idx & 3) * 8;
      const __bf16* src = y + (size_t)(m0 + r) * Cn + k0 + c;
      __builtin_prefetch(src + 32, 0, 3);
      *(v8bf*)&As[r][c] = *(const v8bf*)src;
    }
#endif
    // stage B transposed (fp32 -> bf16 [n][k])
    for (unsigned idx = tid; idx < 32 * 32; idx += 256) {
      unsigned r = idx >> 5, c = (idx & 31) * 4;
      const float* src = w + (size_t)(k0 + r) * Cn + n0 + c;
      const float4 f = *(const float4*)src;
      __builtin_prefetch(src + (size_t)32 * Cn, 0, 3);
      Bs[c + 0][r] = (__bf16)f.x; Bs[c + 1][r] = (__bf16)f.y;
      Bs[c + 2][r] = (__bf16)f.z; Bs[c + 3][r] = (__bf16)f.w;
    }
#if HAVE_TDM
    if (wave == 0) __builtin_amdgcn_s_wait_tensorcnt(0);
#endif
    __syncthreads();

    const v16bf a0 = make_afrag(&As[mg * 32 + nloc][0], kchunk);
    const v16bf a1 = make_afrag(&As[mg * 32 + 16 + nloc][0], kchunk);
#pragma unroll
    for (int j = 0; j < 4; ++j) {
      const v16bf b = make_bfrag(&Bs[ng * 64 + j * 16 + nloc][half << 4]);
      acc[0][j] = WMMA_BF16(a0, b, acc[0][j]);
      acc[1][j] = WMMA_BF16(a1, b, acc[1][j]);
    }
    __syncthreads();
  }

#pragma unroll
  for (unsigned j = 0; j < 4; ++j) {
    const unsigned n = n0 + ng * 64 + j * 16 + nloc;
    const float bn = bias[n];
#pragma unroll
    for (unsigned i = 0; i < 2; ++i) {
#pragma unroll
      for (unsigned r = 0; r < 8; ++r) {
        const unsigned m = m0 + mg * 32 + i * 16 + r + 8 * half;
        out[(size_t)m * Cn + n] = acc[i][j][r] + bn;
      }
    }
  }
}

// ---------------------------------------------------------------------------
extern "C" void kernel_launch(void* const* d_in, const int* in_sizes, int n_in,
                              void* d_out, int out_size, void* d_ws, size_t ws_size,
                              hipStream_t stream) {
  const float* x      = (const float*)d_in[0];
  const float* w_qkv  = (const float*)d_in[1];
  const float* w_proj = (const float*)d_in[2];
  const float* b_proj = (const float*)d_in[3];
  float* out = (float*)d_out;

  const size_t perBuf = (size_t)Bn * Hn * Tn * Dn;  // elements (bf16)
  __bf16* qb = (__bf16*)d_ws;
  __bf16* kb = qb + perBuf;
  __bf16* vt = kb + perBuf;                          // [B,H,D,T]
  __bf16* yb = vt + perBuf;                          // [B,T,C]

  qkv_gemm<<<dim3(N_QKV / 128, Mrows / 128), 256, 0, stream>>>(x, w_qkv, qb, kb, vt);
  attn<<<dim3(Tn / 128, Bn * Hn), 256, 0, stream>>>(qb, kb, vt, yb);
  out_gemm<<<dim3(Cn / 128, Mrows / 128), 256, 0, stream>>>(yb, w_proj, b_proj, out);
}
